// FANet_Structural_74577812128604
// MI455X (gfx1250) — matrix operand order repaired and checked by reading, not compiled, not run on hardware
//
#include <hip/hip_runtime.h>
#include <hip/hip_bf16.h>

typedef __attribute__((ext_vector_type(16))) _Float16 v16h;
typedef __attribute__((ext_vector_type(8)))  float    v8f;

#define H64   64
#define FIN   128
#define NB    256   // graphs
#define NEG_HUGE (-3.402823466e+38f)

// ---------------------------------------------------------------- utilities
__global__ void k_fill(float* __restrict__ p, int n, float v) {
    int i = blockIdx.x * blockDim.x + threadIdx.x;
    if (i < n) p[i] = v;
}

__device__ __forceinline__ void atomicMaxFloat(float* addr, float val) {
    unsigned* ua = (unsigned*)addr;
    unsigned old = __atomic_load_n(ua, __ATOMIC_RELAXED);
    while (__uint_as_float(old) < val) {
        unsigned assumed = old;
        old = atomicCAS(ua, assumed, __float_as_uint(val));
        if (old == assumed) break;
    }
}

// ---------------------------------------------------------------- degree / norm
__global__ void k_deg(const int* __restrict__ dst, float* __restrict__ deg, int E) {
    int e = blockIdx.x * blockDim.x + threadIdx.x;
    if (e < E) atomicAdd(&deg[dst[e]], 1.0f);
}

__global__ void k_dis(float* __restrict__ deg, int N) {
    int i = blockIdx.x * blockDim.x + threadIdx.x;
    if (i < N) {
        float d = deg[i];
        deg[i] = (d > 0.0f) ? rsqrtf(fmaxf(d, 1.0f)) : 0.0f;
    }
}

// ---------------------------------------------------------------- B-fragment prepack (nn1_w, [128x64])
// wpack entry (col, ks, lane): 16 contiguous halfs = lane's v16h fragment.
// packed B order (contiguous k, same as A): lane&15 = N column, lane>=16 selects K+16; element e -> K = ks*32 + 16*(lane>>4) + e
__global__ void k_prepack(const float* __restrict__ w, _Float16* __restrict__ wpack) {
    int t = blockIdx.x * blockDim.x + threadIdx.x;   // 0..511
    if (t >= 512) return;
    int col  = t >> 7;
    int ks   = (t >> 5) & 3;
    int lane = t & 31;
    int n    = col * 16 + (lane & 15);
    int kh   = (lane >> 4) * 16;
    _Float16* o = wpack + (size_t)t * 16;
#pragma unroll
    for (int e = 0; e < 16; ++e) {
        int K = ks * 32 + kh + e;
        o[e] = (_Float16)w[K * H64 + n];
    }
}

// head weight prepack (att1_w, [64x16]): 2 ksteps x 32 lanes
__global__ void k_prepack_head(const float* __restrict__ w, _Float16* __restrict__ wpack) {
    int t = blockIdx.x * blockDim.x + threadIdx.x;   // 0..63
    if (t >= 64) return;
    int ks   = t >> 5;
    int lane = t & 31;
    int n    = lane & 15;
    int kh   = (lane >> 4) * 16;
    _Float16* o = wpack + (size_t)t * 16;
#pragma unroll
    for (int e = 0; e < 16; ++e) {
        int K = ks * 32 + kh + e;
        o[e] = (_Float16)w[K * 16 + n];
    }
}

// ---------------------------------------------------------------- WMMA GEMM: x0 = relu(x @ W + b)
// one block = 32 rows (2 row-tiles) x all 4 col-tiles; 8 waves, one 16x16 tile each.
__global__ void k_gemm_wmma(const float* __restrict__ x, const _Float16* __restrict__ wpack,
                            const float* __restrict__ bias, float* __restrict__ x0, int N) {
    __shared__ _Float16 As[32 * FIN];   // 8 KB
    const int tid = threadIdx.x;
    const size_t base = (size_t)blockIdx.x * 32 * FIN;
    const size_t lim  = (size_t)N * FIN - 1;

    // prefetch next block's A tile (16 KB = 256 threads x 64 B) -> global_prefetch_b8
    if (base + 32 * FIN <= lim)
        __builtin_prefetch(x + base + 32 * FIN + (size_t)tid * 16, 0, 1);

#pragma unroll
    for (int i = 0; i < 16; ++i) {
        int idx = tid + i * 256;
        size_t g = base + idx; if (g > lim) g = lim;
        As[idx] = (_Float16)x[g];
    }
    __syncthreads();

    const int lane = tid & 31, wave = tid >> 5;
    const int rt   = wave >> 2;          // row-tile in block (0/1)
    const int col  = wave & 3;           // col-tile (0..3)
    const int hsel = lane >> 4;
    const int row  = rt * 16 + (lane & 15);

    const unsigned* As2 = (const unsigned*)As;     // half-pairs
    const v16h* bp = (const v16h*)wpack;

    v8f acc = {};
#pragma unroll
    for (int ks = 0; ks < 4; ++ks) {
        union { v16h v; unsigned u[8]; } af;
        const int kb = ks * 32 + hsel * 16;
#pragma unroll
        for (int g = 0; g < 2; ++g)
#pragma unroll
            for (int i = 0; i < 4; ++i)
                af.u[g * 4 + i] = As2[(row * FIN + kb + g * 8 + 2 * i) >> 1];
        v16h bf = bp[(col * 4 + ks) * 32 + lane];
        acc = __builtin_amdgcn_wmma_f32_16x16x32_f16(false, af.v, false, bf,
                                                     (short)0, acc, false, false);
    }

    const int n = lane & 15;
    const float b = bias[col * 16 + n];
    const long rowbase = (long)blockIdx.x * 32 + rt * 16;
    if (rowbase + 16 <= (long)N) {
        // full tile: straight-line coalesced stores
        float* o = x0 + (rowbase + (long)hsel * 8) * H64 + col * 16 + n;
#pragma unroll
        for (int r = 0; r < 8; ++r) {
            float v = acc[r] + b;
            o[(long)r * H64] = v > 0.0f ? v : 0.0f;
        }
    } else {
#pragma unroll
        for (int r = 0; r < 8; ++r) {
            long m = rowbase + r + hsel * 8;
            if (m < (long)N) {
                float v = acc[r] + b;
                x0[m * H64 + col * 16 + n] = v > 0.0f ? v : 0.0f;
            }
        }
    }
}

// ---------------------------------------------------------------- per-node dots: hl = h.wl, hr = h.wr
__global__ void k_dots(const float* __restrict__ h, const float* __restrict__ wl,
                       const float* __restrict__ wr, float* __restrict__ hl,
                       float* __restrict__ hr, int N) {
    int node = (blockIdx.x * blockDim.x + threadIdx.x) >> 5;
    int lane = threadIdx.x & 31;
    if (node >= N) return;
    const float* hp = h + (size_t)node * H64;
    float v0 = hp[lane], v1 = hp[lane + 32];
    float sl = v0 * wl[lane] + v1 * wl[lane + 32];
    float sr = v0 * wr[lane] + v1 * wr[lane + 32];
#pragma unroll
    for (int off = 16; off; off >>= 1) {
        sl += __shfl_xor(sl, off, 32);
        sr += __shfl_xor(sr, off, 32);
    }
    if (lane == 0) { hl[node] = sl; hr[node] = sr; }
}

// ---------------------------------------------------------------- h_out = eps * x0 (residual pre-init)
__global__ void k_init_eps(float* __restrict__ hout, const float* __restrict__ x0, float eps, int n) {
    int i = blockIdx.x * blockDim.x + threadIdx.x;
    if (i < n) hout[i] = eps * x0[i];
}

// ---------------------------------------------------------------- edge scatter: wave per edge
__global__ void k_edge(const int* __restrict__ src, const int* __restrict__ dst,
                       const float* __restrict__ hl, const float* __restrict__ hr,
                       const float* __restrict__ dis, const float* __restrict__ hin,
                       float* __restrict__ hout, int E) {
    int e = (blockIdx.x * blockDim.x + threadIdx.x) >> 5;
    int lane = threadIdx.x & 31;
    if (e >= E) return;
    int s = src[e], d = dst[e];
    float a = tanhf(hl[s] + hr[d]);
    float w = a * dis[s] * dis[d];
    const float* hs = hin + (size_t)s * H64;
    float* ho = hout + (size_t)d * H64;
    atomicAdd(&ho[lane],      w * hs[lane]);
    atomicAdd(&ho[lane + 32], w * hs[lane + 32]);
}

// ---------------------------------------------------------------- closeness scores
__global__ void k_scores(const float* __restrict__ c, const float* __restrict__ cw,
                         const float* __restrict__ cb, float* __restrict__ scores, int N) {
    int i = blockIdx.x * blockDim.x + threadIdx.x;
    if (i >= N) return;
    const float* cp = c + (size_t)i * 8;
    float s = cb[0];
#pragma unroll
    for (int k = 0; k < 8; ++k) s += cp[k] * cw[k];
    scores[i] = s;
}

__global__ void k_smax(const float* __restrict__ scores, const int* __restrict__ batch,
                       float* __restrict__ smax, int N) {
    int i = blockIdx.x * blockDim.x + threadIdx.x;
    if (i < N) atomicMaxFloat(&smax[batch[i]], scores[i]);
}

__global__ void k_expsum(const float* __restrict__ scores, const int* __restrict__ batch,
                         const float* __restrict__ smax, float* __restrict__ den,
                         float* __restrict__ cnt, float* __restrict__ ex, int N) {
    int i = blockIdx.x * blockDim.x + threadIdx.x;
    if (i >= N) return;
    int b = batch[i];
    float e = expf(scores[i] - smax[b]);
    ex[i] = e;
    atomicAdd(&den[b], e);
    atomicAdd(&cnt[b], 1.0f);
}

__global__ void k_pfinal(float* __restrict__ p, const int* __restrict__ batch,
                         const float* __restrict__ den, const float* __restrict__ cnt, int N) {
    int i = blockIdx.x * blockDim.x + threadIdx.x;
    if (i >= N) return;
    int b = batch[i];
    p[i] = p[i] / den[b] * cnt[b];
}

__global__ void k_pool(const float* __restrict__ p, const float* __restrict__ h,
                       const int* __restrict__ batch, float* __restrict__ pooled, int n) {
    int i = blockIdx.x * blockDim.x + threadIdx.x;
    if (i >= n) return;
    int node = i >> 6, f = i & 63;
    float v = p[node] * h[i];
    atomicMaxFloat(&pooled[batch[node] * H64 + f], v);
}

// ---------------------------------------------------------------- head via WMMA:
// out = relu(pooled[256,64] @ att1_w[64,16] + b1) @ att2_w[16,1] + b2
// one wave per 16-graph tile; K=64 -> 2 WMMAs; contraction over the 16 hidden
// units done with a 16-lane shfl_xor reduction in the epilogue.
__global__ void k_head_wmma(const float* __restrict__ pooled, const _Float16* __restrict__ wpack2,
                            const float* __restrict__ b1, const float* __restrict__ w2,
                            const float* __restrict__ b2, float* __restrict__ out) {
    const int lane = threadIdx.x & 31, wave = threadIdx.x >> 5;
    const int rowTile = blockIdx.x * 8 + wave;       // 0..15 (256 graphs / 16)
    const int hsel = lane >> 4, n = lane & 15;
    const int row = rowTile * 16 + n;                // graph index for A rows

    const v16h* bp = (const v16h*)wpack2;
    v8f acc = {};
#pragma unroll
    for (int ks = 0; ks < 2; ++ks) {
        union { v16h v; _Float16 h[16]; } af;
#pragma unroll
        for (int g = 0; g < 2; ++g)
#pragma unroll
            for (int i = 0; i < 8; ++i) {
                int e = g * 8 + i;
                int K = ks * 32 + hsel * 16 + g * 8 + i;
                af.h[e] = (_Float16)pooled[row * H64 + K];
            }
        v16h bf = bp[ks * 32 + lane];
        acc = __builtin_amdgcn_wmma_f32_16x16x32_f16(false, af.v, false, bf,
                                                     (short)0, acc, false, false);
    }

    const float w2n = w2[n], b1n = b1[n], bout = b2[0];
#pragma unroll
    for (int r = 0; r < 8; ++r) {
        float s = acc[r] + b1n;              // hidden unit n for graph m
        s = (s > 0.0f ? s : 0.0f) * w2n;
#pragma unroll
        for (int m = 8; m; m >>= 1) s += __shfl_xor(s, m, 32);  // reduce 16 lanes
        if (n == 0) out[rowTile * 16 + r + hsel * 8] = s + bout;
    }
}

// ================================================================ launch
extern "C" void kernel_launch(void* const* d_in, const int* in_sizes, int n_in,
                              void* d_out, int out_size, void* d_ws, size_t ws_size,
                              hipStream_t stream) {
    const float* x      = (const float*)d_in[0];
    const float* clos   = (const float*)d_in[1];
    const int*   eidx   = (const int*)  d_in[2];
    const int*   batch  = (const int*)  d_in[3];
    const float* nn1_w  = (const float*)d_in[4];
    const float* nn1_b  = (const float*)d_in[5];
    const float* attL[3]= {(const float*)d_in[6], (const float*)d_in[8],  (const float*)d_in[10]};
    const float* attR[3]= {(const float*)d_in[7], (const float*)d_in[9],  (const float*)d_in[11]};
    const float* cls_w  = (const float*)d_in[12];
    const float* cls_b  = (const float*)d_in[13];
    const float* att1_w = (const float*)d_in[14];
    const float* att1_b = (const float*)d_in[15];
    const float* att2_w = (const float*)d_in[16];
    const float* att2_b = (const float*)d_in[17];
    float* out = (float*)d_out;

    const int N = in_sizes[0] / FIN;
    const int E = in_sizes[2] / 2;
    const int* src = eidx;
    const int* dst = eidx + E;

    // workspace carve-out (256B aligned)
    size_t off = 0;
    auto take = [&](size_t bytes) {
        void* p = (char*)d_ws + off;
        off += (bytes + 255) & ~(size_t)255;
        return p;
    };
    float*    x0     = (float*)take((size_t)N * H64 * 4);
    float*    hA     = (float*)take((size_t)N * H64 * 4);
    float*    hB     = (float*)take((size_t)N * H64 * 4);
    float*    dis    = (float*)take((size_t)N * 4);
    float*    hl     = (float*)take((size_t)N * 4);
    float*    hr     = (float*)take((size_t)N * 4);
    float*    scores = (float*)take((size_t)N * 4);
    float*    pbuf   = (float*)take((size_t)N * 4);
    float*    smax   = (float*)take(NB * 4);
    float*    den    = (float*)take(NB * 4);
    float*    cnt    = (float*)take(NB * 4);
    float*    pooled = (float*)take(NB * H64 * 4);
    _Float16* wpack  = (_Float16*)take(512 * 16 * 2);
    _Float16* wpack2 = (_Float16*)take(64 * 16 * 2);

    const int T = 256;
    auto blk = [](long n, int t) { return (unsigned)((n + t - 1) / t); };

    // degree -> dis = deg>0 ? rsqrt(max(deg,1)) : 0
    k_fill<<<blk(N, T), T, 0, stream>>>(dis, N, 0.0f);
    k_deg <<<blk(E, T), T, 0, stream>>>(dst, dis, E);
    k_dis <<<blk(N, T), T, 0, stream>>>(dis, N);

    // x0 = relu(x @ nn1_w + b)   (WMMA f16->f32)
    k_prepack     <<<2, T, 0, stream>>>(nn1_w, wpack);
    k_prepack_head<<<1, T, 0, stream>>>(att1_w, wpack2);
    k_gemm_wmma   <<<blk(N, 32), T, 0, stream>>>(x, wpack, nn1_b, x0, N);

    // 3 FAConv layers: x0 -> hA -> hB -> hA
    const float* hin = x0;
    float* hout[3] = {hA, hB, hA};
    for (int L = 0; L < 3; ++L) {
        k_dots    <<<blk((long)N * 32, T), T, 0, stream>>>(hin, attL[L], attR[L], hl, hr, N);
        k_init_eps<<<blk((long)N * H64, T), T, 0, stream>>>(hout[L], x0, 0.1f, N * H64);
        k_edge    <<<blk((long)E * 32, T), T, 0, stream>>>(src, dst, hl, hr, dis, hin, hout[L], E);
        hin = hout[L];
    }

    // per-graph softmax(closeness) * count, weighted max-pool
    k_scores<<<blk(N, T), T, 0, stream>>>(clos, cls_w, cls_b, scores, N);
    k_fill  <<<1, T, 0, stream>>>(smax, NB, NEG_HUGE);
    k_fill  <<<1, T, 0, stream>>>(den,  NB, 0.0f);
    k_fill  <<<1, T, 0, stream>>>(cnt,  NB, 0.0f);
    k_smax  <<<blk(N, T), T, 0, stream>>>(scores, batch, smax, N);
    k_expsum<<<blk(N, T), T, 0, stream>>>(scores, batch, smax, den, cnt, pbuf, N);
    k_pfinal<<<blk(N, T), T, 0, stream>>>(pbuf, batch, den, cnt, N);
    k_fill  <<<blk(NB * H64, T), T, 0, stream>>>(pooled, NB * H64, NEG_HUGE);
    k_pool  <<<blk((long)N * H64, T), T, 0, stream>>>(pbuf, hin, batch, pooled, N * H64);

    // head MLP (WMMA)
    k_head_wmma<<<2, T, 0, stream>>>(pooled, wpack2, att1_b, att2_w, att2_b, out);
}
